// XLNetViT_42777874268827
// MI455X (gfx1250) — matrix operand order repaired and checked by reading, not compiled, not run on hardware
//
#include <hip/hip_runtime.h>
#include <math.h>

// ---------------------------------------------------------------------------
// CDNA5 (gfx1250) XLNet-ViT forward.  wave32, WMMA bf16 (16x16x32, f32 acc),
// async global->LDS staging (ASYNCcnt) with double-buffered LDS in the GEMM.
// ---------------------------------------------------------------------------

typedef __attribute__((ext_vector_type(16))) __bf16 v16bf;
typedef __attribute__((ext_vector_type(8)))  float  v8f;

union Frag { v16bf v; uint4 q[2]; };

#define DEV __device__ __forceinline__

DEV unsigned short f2bf(float f) {              // RNE float -> bf16 bits
  unsigned u = __float_as_uint(f);
  u += 0x7FFFu + ((u >> 16) & 1u);
  return (unsigned short)(u >> 16);
}

static constexpr int Bc = 8, Lc = 196, Sc = 343, Dc = 768, Hc = 12, HDc = 64;
static constexpr int NSEEN = 49, NTGT = 147;
static constexpr int Mtot = Bc * Sc;            // 2744

DEV v8f vzero8() { v8f z = {0.f,0.f,0.f,0.f,0.f,0.f,0.f,0.f}; return z; }

DEV v8f wmma_bf16(v16bf a, v16bf b, v8f c) {
  return __builtin_amdgcn_wmma_f32_16x16x32_bf16(false, a, false, b,
                                                 (short)0, c, false, false);
}

// Wave-relative LDS byte offset of a __shared__ object (flat low 32 bits).
DEV unsigned lds_off32(const void* p) { return (unsigned)(size_t)p; }

// Issue this thread's 4 async 16B transfers for one 32-wide K tile:
// 2 chunks of A (+0,+16B) and 2 chunks of W.  INST_OFFSET applies to both
// the global and the LDS address, so one VGPR pair + one LDS offset each.
DEV void stage_async(const unsigned short* ga, unsigned offA,
                     const unsigned short* gw, unsigned offB) {
  asm volatile(
      "global_load_async_to_lds_b128 %0, %1, off\n\t"
      "global_load_async_to_lds_b128 %0, %1, off offset:16\n\t"
      "global_load_async_to_lds_b128 %2, %3, off\n\t"
      "global_load_async_to_lds_b128 %2, %3, off offset:16"
      :: "v"(offA), "v"((const void*)ga), "v"(offB), "v"((const void*)gw)
      : "memory");
}

DEV void wait_async4() { asm volatile("s_wait_asynccnt 0x4" ::: "memory"); }
DEV void wait_async0() { asm volatile("s_wait_asynccnt 0x0" ::: "memory"); }

// ---------------------------------------------------------------------------
// Generic GEMM:  C[M,N] = A[M,K](bf16) * W[N,K](bf16)^T + bias [+res] [gelu]
// 256 threads = 8 waves; block tile 128x128, K-step 32; wave tile 64x32.
// LDS row stride 40 bf16 (80B): conflict-free b128 fragment reads.
// Double-buffered LDS, filled by GLOBAL_LOAD_ASYNC_TO_LDS_B128.
// ---------------------------------------------------------------------------
template<bool GELU_, bool RES_, bool OUTBF_>
__global__ __launch_bounds__(256)
void gemm_bf16_kernel(const unsigned short* __restrict__ A,
                      const unsigned short* __restrict__ W,
                      const float* __restrict__ bias,
                      const float* res, float* Cf, unsigned short* Cb,
                      int M, int N, int K)
{
  __shared__ __align__(16) unsigned short As[2][128 * 40];
  __shared__ __align__(16) unsigned short Bs[2][128 * 40];

  const int t    = threadIdx.x;
  const int lane = t & 31, wave = t >> 5;
  const int wm   = wave & 1, wn = wave >> 1;      // 2 (M) x 4 (N) waves
  const int nl   = lane & 15, hi = lane >> 4;
  const int m0   = blockIdx.x * 128, n0 = blockIdx.y * 128;
  const int rit  = t >> 1, half = t & 1;          // staging: 2 thr / row

  v8f acc[4][2];
#pragma unroll
  for (int i = 0; i < 4; ++i)
#pragma unroll
    for (int j = 0; j < 2; ++j) acc[i][j] = vzero8();

  int arow = m0 + rit; if (arow > M - 1) arow = M - 1;   // clamp (pad rows)
  const int wrow = n0 + rit;                              // N % 128 == 0

  const unsigned short* ga = A + (size_t)arow * K + half * 16;
  const unsigned short* gw = W + (size_t)wrow * K + half * 16;
  const unsigned offA[2] = { lds_off32(&As[0][rit * 40 + half * 16]),
                             lds_off32(&As[1][rit * 40 + half * 16]) };
  const unsigned offB[2] = { lds_off32(&Bs[0][rit * 40 + half * 16]),
                             lds_off32(&Bs[1][rit * 40 + half * 16]) };

  // prologue: stage tile 0 into buffer 0
  stage_async(ga, offA[0], gw, offB[0]);

  int p = 0;
  for (int k0 = 0; k0 < K; k0 += 32) {
    const bool hasNext = (k0 + 32 < K);
    if (hasNext)                                   // stream next tile into p^1
      stage_async(ga + k0 + 32, offA[p ^ 1], gw + k0 + 32, offB[p ^ 1]);
    // retire only the current tile's 4 transfers; keep next tile in flight
    if (hasNext) wait_async4(); else wait_async0();
    __syncthreads();

    Frag fa[4], fb[2];
#pragma unroll
    for (int mt = 0; mt < 4; ++mt) {        // A 16x32: chunks [kb,kb+8)+[kb+16,kb+24)
      const unsigned short* pp = As[p] + (wm * 64 + mt * 16 + nl) * 40 + hi * 8;
      fa[mt].q[0] = *(const uint4*)pp;
      fa[mt].q[1] = *(const uint4*)(pp + 16);
    }
#pragma unroll
    for (int nt = 0; nt < 2; ++nt) {        // B 32x16: chunks [kb,kb+8)+[kb+16,kb+24)
      const unsigned short* pp = Bs[p] + (wn * 32 + nt * 16 + nl) * 40 + hi * 8;
      fb[nt].q[0] = *(const uint4*)pp;
      fb[nt].q[1] = *(const uint4*)(pp + 16);
    }
#pragma unroll
    for (int mt = 0; mt < 4; ++mt)
#pragma unroll
      for (int nt = 0; nt < 2; ++nt)
        acc[mt][nt] = wmma_bf16(fa[mt].v, fb[nt].v, acc[mt][nt]);
    __syncthreads();                        // all waves done reading buf p
    p ^= 1;
  }

  // Epilogue.  C 16x16 f32: lane -> col nl, rows hi*8+v in VGPR v.
#pragma unroll
  for (int mt = 0; mt < 4; ++mt)
#pragma unroll
    for (int nt = 0; nt < 2; ++nt) {
      const int col = n0 + wn * 32 + nt * 16 + nl;
#pragma unroll
      for (int v = 0; v < 8; ++v) {
        const int row = m0 + wm * 64 + mt * 16 + hi * 8 + v;
        if (row < M) {
          float x = acc[mt][nt][v] + bias[col];
          if (RES_)  x += res[(size_t)row * N + col];
          if (GELU_) x = 0.5f * x * (1.f + erff(x * 0.70710678118654752f));
          if (OUTBF_) Cb[(size_t)row * N + col] = f2bf(x);
          else        Cf[(size_t)row * N + col] = x;
        }
      }
    }
}

// ---------------------------------------------------------------------------
// Attention: one wave32 per (query-16-tile, head, batch).
// scores = softmax(Q K^T * 0.125 + mask);  O = scores V   (all via WMMA)
// ---------------------------------------------------------------------------
__global__ __launch_bounds__(32)
void attn_kernel(const unsigned short* __restrict__ qkv,  // [B,S,2304] bf16
                 unsigned short* __restrict__ out)        // [B,S,768]  bf16
{
  __shared__ __align__(16) unsigned short vT[64 * 232];   // V^T [hd][key] pad
  __shared__ float sc[16 * 228];                          // raw scores
  __shared__ __align__(16) unsigned short pr[16 * 232];   // bf16 probs (unnorm)
  __shared__ float sinv[16];                              // 1/rowsum

  const int lane = threadIdx.x;
  const int q0 = blockIdx.x * 16, h = blockIdx.y, b = blockIdx.z;
  const size_t base = (size_t)b * Sc * 2304;
  const int qoff = h * 64, koff = 768 + h * 64, voff = 1536 + h * 64;
  const int r = lane & 15, hi = lane >> 4;

  // 1) stage V^T to LDS (keys padded/zeroed to 224)
  for (int idx = lane; idx < 224 * 64; idx += 32) {
    int key = idx >> 6, d = idx & 63;
    unsigned short val = 0;
    if (key < Lc) val = qkv[base + (size_t)key * 2304 + voff + d];
    vT[d * 232 + key] = val;
  }

  // 2) Q A-fragments (16x64, two K-steps of 32) straight from global
  const int qg = q0 + r;
  const int qr = (qg < Sc) ? qg : (Sc - 1);
  const unsigned short* qrow = qkv + base + (size_t)qr * 2304 + qoff;
  Frag aq[2];
#pragma unroll
  for (int ks = 0; ks < 2; ++ks) {
    const int kb = ks * 32 + hi * 8;
    aq[ks].q[0] = *(const uint4*)(qrow + kb);
    aq[ks].q[1] = *(const uint4*)(qrow + kb + 16);
  }

  // 3) scores: 13 key tiles of 16, B-frags from K rows in global
  for (int kt = 0; kt < 13; ++kt) {
    int key = kt * 16 + r; if (key > Lc - 1) key = Lc - 1;
    const unsigned short* krow = qkv + base + (size_t)key * 2304 + koff;
    v8f acc = vzero8();
#pragma unroll
    for (int ks = 0; ks < 2; ++ks) {
      const int kb = ks * 32 + hi * 8;
      Frag fb;
      fb.q[0] = *(const uint4*)(krow + kb);
      fb.q[1] = *(const uint4*)(krow + kb + 16);
      acc = wmma_bf16(aq[ks].v, fb.v, acc);
    }
#pragma unroll
    for (int v = 0; v < 8; ++v)
      sc[(hi * 8 + v) * 228 + kt * 16 + r] = acc[v];
  }
  __syncthreads();

  // 4) masked softmax (2 lanes per row; mask rebuilt from NUM_SEEN)
  {
    const int limit = (qg < Lc) ? (qg + 1)
                    : (qg < Sc) ? (NSEEN + (qg - Lc)) : 1;
    const float* srow = sc + r * 228;
    float mx = -3.0e38f;
    for (int c = hi * 112; c < hi * 112 + 112; ++c)
      if (c < limit) { float x = srow[c] * 0.125f; mx = fmaxf(mx, x); }
    mx = fmaxf(mx, __shfl_xor(mx, 16, 32));
    float sum = 0.f;
    for (int c = hi * 112; c < hi * 112 + 112; ++c) {
      float p = 0.f;
      if (c < limit) p = __expf(srow[c] * 0.125f - mx);
      sum += p;
      pr[r * 232 + c] = f2bf(p);
    }
    sum += __shfl_xor(sum, 16, 32);
    if (hi == 0) sinv[r] = 1.f / sum;
  }
  __syncthreads();

  // 5) O = P V : 4 hd-tiles, 7 K-steps of 32 padded keys
  v8f av[4];
#pragma unroll
  for (int nt = 0; nt < 4; ++nt) av[nt] = vzero8();
  for (int ks = 0; ks < 7; ++ks) {
    Frag fp;
    const unsigned short* prow = pr + r * 232 + ks * 32 + hi * 8;
    fp.q[0] = *(const uint4*)prow;
    fp.q[1] = *(const uint4*)(prow + 16);
#pragma unroll
    for (int nt = 0; nt < 4; ++nt) {
      Frag fv;
      const unsigned short* vrow = vT + (nt * 16 + r) * 232 + ks * 32 + hi * 8;
      fv.q[0] = *(const uint4*)vrow;
      fv.q[1] = *(const uint4*)(vrow + 16);
      av[nt] = wmma_bf16(fp.v, fv.v, av[nt]);
    }
  }
#pragma unroll
  for (int nt = 0; nt < 4; ++nt)
#pragma unroll
    for (int v = 0; v < 8; ++v) {
      const int m = hi * 8 + v, qg3 = q0 + m;
      if (qg3 < Sc)
        out[((size_t)b * Sc + qg3) * Dc + h * 64 + nt * 16 + r] =
            f2bf(av[nt][v] * sinv[m]);
    }
}

// ---------------------------------------------------------------------------
// LayerNorm: one 256-thread WG per row of 768, f32 in -> bf16 out
// ---------------------------------------------------------------------------
__global__ __launch_bounds__(256)
void ln_kernel(const float* __restrict__ x, const float* __restrict__ w,
               const float* __restrict__ b, unsigned short* __restrict__ y)
{
  __shared__ float sbuf[256];
  const int row = blockIdx.x, t = threadIdx.x;
  const float* xr = x + (size_t)row * Dc;
  float v0 = xr[t], v1 = xr[t + 256], v2 = xr[t + 512];
  sbuf[t] = v0 + v1 + v2; __syncthreads();
  for (int o = 128; o > 0; o >>= 1) { if (t < o) sbuf[t] += sbuf[t + o]; __syncthreads(); }
  const float mean = sbuf[0] * (1.f / 768.f);
  __syncthreads();
  const float d0 = v0 - mean, d1 = v1 - mean, d2 = v2 - mean;
  sbuf[t] = d0 * d0 + d1 * d1 + d2 * d2; __syncthreads();
  for (int o = 128; o > 0; o >>= 1) { if (t < o) sbuf[t] += sbuf[t + o]; __syncthreads(); }
  const float rstd = rsqrtf(sbuf[0] * (1.f / 768.f) + 1e-5f);
  unsigned short* yr = y + (size_t)row * Dc;
  yr[t]       = f2bf(d0 * rstd * w[t]       + b[t]);
  yr[t + 256] = f2bf(d1 * rstd * w[t + 256] + b[t + 256]);
  yr[t + 512] = f2bf(d2 * rstd * w[t + 512] + b[t + 512]);
}

// ---------------------------------------------------------------------------
// Small elementwise kernels
// ---------------------------------------------------------------------------
__global__ void f32_to_bf16_kernel(const float* __restrict__ in,
                                   unsigned short* __restrict__ out, int n) {
  int i = blockIdx.x * 256 + threadIdx.x;
  if (i < n) out[i] = f2bf(in[i]);
}

__global__ void patchify_kernel(const float* __restrict__ imgs,
                                unsigned short* __restrict__ out) {
  int i = blockIdx.x * 256 + threadIdx.x;
  if (i >= Bc * Lc * Dc) return;
  int row = i / Dc, col = i % Dc;
  int b = row / Lc, l = row % Lc;
  int gh = l / 14, gw = l % 14;
  int c = col >> 8, rem = col & 255, p = rem >> 4, q = rem & 15;
  size_t src = (((size_t)(b * 3 + c) * 224) + gh * 16 + p) * 224 + gw * 16 + q;
  out[i] = f2bf(imgs[src]);
}

__global__ void build_seq_kernel(const float* __restrict__ pg,
                                 const float* __restrict__ pos,
                                 const float* __restrict__ mtok,
                                 float* __restrict__ hbuf) {
  int i = blockIdx.x * 256 + threadIdx.x;
  if (i >= Bc * Sc * Dc) return;
  int d = i % Dc, rs = i / Dc;
  int s = rs % Sc, b = rs / Sc;
  float v;
  if (s < Lc) v = pg[((size_t)b * Lc + s) * Dc + d] + pos[(size_t)(1 + s) * Dc + d];
  else        v = pos[(size_t)(1 + NSEEN + (s - Lc)) * Dc + d] + mtok[d];
  hbuf[i] = v;
}

__global__ void gather_query_kernel(const unsigned short* __restrict__ yall,
                                    unsigned short* __restrict__ g) {
  int i = blockIdx.x * 256 + threadIdx.x;
  if (i >= Bc * NTGT * Dc) return;
  int d = i % Dc, rt = i / Dc;
  int b = rt / NTGT, tq = rt % NTGT;
  g[i] = yall[((size_t)b * Sc + Lc + tq) * Dc + d];
}

// ---------------------------------------------------------------------------
// Host orchestration
// ---------------------------------------------------------------------------
static inline int ceil_div(int a, int b) { return (a + b - 1) / b; }

extern "C" void kernel_launch(void* const* d_in, const int* in_sizes, int n_in,
                              void* d_out, int out_size, void* d_ws, size_t ws_size,
                              hipStream_t stream) {
  (void)in_sizes; (void)n_in; (void)out_size; (void)ws_size;
  const float* imgs   = (const float*)d_in[0];
  const float* pos    = (const float*)d_in[1];
  const float* mtok   = (const float*)d_in[2];
  const float* conv_w = (const float*)d_in[3];
  const float* conv_b = (const float*)d_in[4];
  const float* qkv_w  = (const float*)d_in[5];
  const float* qkv_b  = (const float*)d_in[6];
  const float* proj_w = (const float*)d_in[7];
  const float* proj_b = (const float*)d_in[8];
  const float* ln1_w  = (const float*)d_in[9];
  const float* ln1_b  = (const float*)d_in[10];
  const float* ln2_w  = (const float*)d_in[11];
  const float* ln2_b  = (const float*)d_in[12];
  const float* fc1_w  = (const float*)d_in[13];
  const float* fc1_b  = (const float*)d_in[14];
  const float* fc2_w  = (const float*)d_in[15];
  const float* fc2_b  = (const float*)d_in[16];
  const float* norm_w = (const float*)d_in[17];
  const float* norm_b = (const float*)d_in[18];
  const float* head_w = (const float*)d_in[19];
  const float* head_b = (const float*)d_in[20];

  char* ws = (char*)d_ws;
  auto alloc = [&](size_t bytes) {
    char* p = ws; ws += (bytes + 255) & ~(size_t)255; return p;
  };
  float*          hbuf = (float*)         alloc((size_t)Mtot * Dc * 4);
  unsigned short* ybuf = (unsigned short*)alloc((size_t)Mtot * Dc * 2);
  unsigned short* qkvb = (unsigned short*)alloc((size_t)Mtot * 2304 * 2);
  unsigned short* aout = (unsigned short*)alloc((size_t)Mtot * Dc * 2);
  unsigned short* mlpb = (unsigned short*)alloc((size_t)Mtot * 3072 * 2);
  unsigned short* wbuf = (unsigned short*)alloc((size_t)3072 * 768 * 2);
  float*          pgout = (float*)mlpb;          // alias: pre-loop only
  unsigned short* gbuf  = aout;                  // alias: post-loop only

  const dim3 blk(256);

  // ---- patch embed + sequence build -------------------------------------
  f32_to_bf16_kernel<<<ceil_div(768 * 768, 256), blk, 0, stream>>>(conv_w, wbuf, 768 * 768);
  patchify_kernel<<<ceil_div(Bc * Lc * Dc, 256), blk, 0, stream>>>(imgs, ybuf);
  gemm_bf16_kernel<false, false, false><<<dim3(ceil_div(Bc * Lc, 128), 6), blk, 0, stream>>>(
      ybuf, wbuf, conv_b, nullptr, pgout, nullptr, Bc * Lc, 768, 768);
  build_seq_kernel<<<ceil_div(Bc * Sc * Dc, 256), blk, 0, stream>>>(pgout, pos, mtok, hbuf);

  // ---- transformer blocks ------------------------------------------------
  for (int l = 0; l < 12; ++l) {
    ln_kernel<<<Mtot, blk, 0, stream>>>(hbuf, ln1_w + l * 768, ln1_b + l * 768, ybuf);
    f32_to_bf16_kernel<<<ceil_div(2304 * 768, 256), blk, 0, stream>>>(
        qkv_w + (size_t)l * 2304 * 768, wbuf, 2304 * 768);
    gemm_bf16_kernel<false, false, true><<<dim3(ceil_div(Mtot, 128), 18), blk, 0, stream>>>(
        ybuf, wbuf, qkv_b + l * 2304, nullptr, nullptr, qkvb, Mtot, 2304, 768);
    attn_kernel<<<dim3(ceil_div(Sc, 16), Hc, Bc), dim3(32), 0, stream>>>(qkvb, aout);
    f32_to_bf16_kernel<<<ceil_div(768 * 768, 256), blk, 0, stream>>>(
        proj_w + (size_t)l * 768 * 768, wbuf, 768 * 768);
    gemm_bf16_kernel<false, true, false><<<dim3(ceil_div(Mtot, 128), 6), blk, 0, stream>>>(
        aout, wbuf, proj_b + l * 768, hbuf, hbuf, nullptr, Mtot, 768, 768);
    ln_kernel<<<Mtot, blk, 0, stream>>>(hbuf, ln2_w + l * 768, ln2_b + l * 768, ybuf);
    f32_to_bf16_kernel<<<ceil_div(3072 * 768, 256), blk, 0, stream>>>(
        fc1_w + (size_t)l * 3072 * 768, wbuf, 3072 * 768);
    gemm_bf16_kernel<true, false, true><<<dim3(ceil_div(Mtot, 128), 24), blk, 0, stream>>>(
        ybuf, wbuf, fc1_b + l * 3072, nullptr, nullptr, mlpb, Mtot, 3072, 768);
    f32_to_bf16_kernel<<<ceil_div(768 * 3072, 256), blk, 0, stream>>>(
        fc2_w + (size_t)l * 768 * 3072, wbuf, 768 * 3072);
    gemm_bf16_kernel<false, true, false><<<dim3(ceil_div(Mtot, 128), 6), blk, 0, stream>>>(
        mlpb, wbuf, fc2_b + l * 768, hbuf, hbuf, nullptr, Mtot, 768, 3072);
  }

  // ---- final norm + head on query stream --------------------------------
  ln_kernel<<<Mtot, blk, 0, stream>>>(hbuf, norm_w, norm_b, ybuf);
  gather_query_kernel<<<ceil_div(Bc * NTGT * Dc, 256), blk, 0, stream>>>(ybuf, gbuf);
  f32_to_bf16_kernel<<<ceil_div(768 * 768, 256), blk, 0, stream>>>(head_w, wbuf, 768 * 768);
  gemm_bf16_kernel<false, false, false><<<dim3(ceil_div(Bc * NTGT, 128), 6), blk, 0, stream>>>(
      gbuf, wbuf, head_b, nullptr, (float*)d_out, nullptr, Bc * NTGT, 768, 768);
}
